// LiGRU_30477087933281
// MI455X (gfx1250) — matrix operand order repaired and checked by reading, not compiled, not run on hardware
//
#include <hip/hip_runtime.h>

// ---------------------------------------------------------------- types
typedef __attribute__((ext_vector_type(16))) __bf16 v16bf;
typedef __attribute__((ext_vector_type(8)))  float  v8f;

union BfFrag { v16bf v; uint4 q[2]; };

#define BB   16
#define TT   1500
#define FF   80
#define KF   96      // F padded to multiple of 32
#define HH   512
#define G2   1024    // 2*H

// ws layout (byte offsets, all 256B aligned)
#define OFF_WBUF  0ull                       // 24000*1024 bf16 = 49,152,000
#define OFF_H0    49152000ull                // 24000*512  bf16 = 24,576,000
#define OFF_XBF   73728000ull                // 24000*96   bf16 =  4,608,000
#define OFF_W0BF  78336000ull                // 1024*96    bf16 =    196,608
#define OFF_U0BF  78532608ull                // 1024*512   bf16 =  1,048,576
#define OFF_W1BF  79581184ull
#define OFF_U1BF  80629760ull
#define OFF_HX0   81678336ull                // 2*16*512 bf16 = 32,768 (double buffer)
#define OFF_HX1   81711104ull
#define OFF_BAR   81743872ull

// ------------------------------------------------- WMMA fragment loaders
// A-matrix 16x32 bf16 (ISA 7.12.2): lane l<16 -> row l; elems0-7 = K k0+8*hi..,
// elems8-15 = K k0+16+8*hi..  (two 16B row-major loads per lane)
__device__ __forceinline__ v16bf load_a_frag(const __bf16* rowptr, int k0) {
  const int hi = (threadIdx.x >> 4) & 1;
  BfFrag f;
  f.q[0] = *(const uint4*)(rowptr + k0 + hi * 8);
  f.q[1] = *(const uint4*)(rowptr + k0 + 16 + hi * 8);
  return f.v;
}
// B-matrix 32x16 bf16, B[k][n] = U[n][k]: lane -> column n = lane&15,
// elems0-7 = K k0+8*hi.., elems8-15 = K k0+16+8*hi.. along U row n.
__device__ __forceinline__ v16bf load_b_frag(const __bf16* rowptr, int k0) {
  const int hi = (threadIdx.x >> 4) & 1;
  BfFrag f;
  f.q[0] = *(const uint4*)(rowptr + k0 + hi * 8);
  f.q[1] = *(const uint4*)(rowptr + k0 + 16 + hi * 8);
  return f.v;
}

__device__ __forceinline__ v8f wmma_bf16(v16bf a, v16bf b, v8f c) {
  return __builtin_amdgcn_wmma_f32_16x16x32_bf16(false, a, false, b,
                                                 (short)0, c, false, false);
}

__device__ __forceinline__ v8f vzero8() {
  v8f z;
#pragma unroll
  for (int i = 0; i < 8; ++i) z[i] = 0.0f;
  return z;
}

// ---------------------------------------------------------- prep kernel
__global__ __launch_bounds__(256)
void k_prep(const float* __restrict__ x,  const float* __restrict__ W0,
            const float* __restrict__ U0, const float* __restrict__ W1,
            const float* __restrict__ U1,
            __bf16* __restrict__ xbf,  __bf16* __restrict__ W0bf,
            __bf16* __restrict__ U0bf, __bf16* __restrict__ W1bf,
            __bf16* __restrict__ U1bf,
            __bf16* __restrict__ hx0,  __bf16* __restrict__ hx1,
            unsigned* __restrict__ bar) {
  const size_t tid = (size_t)blockIdx.x * blockDim.x + threadIdx.x;
  const size_t stride = (size_t)gridDim.x * blockDim.x;
  const size_t NX = (size_t)BB * TT * KF;
  for (size_t i = tid; i < NX; i += stride) {
    int k = (int)(i % KF); size_t r = i / KF;
    xbf[i] = (k < FF) ? (__bf16)x[r * FF + k] : (__bf16)0.0f;
  }
  const size_t NW0 = (size_t)G2 * KF;
  for (size_t i = tid; i < NW0; i += stride) {
    int k = (int)(i % KF); size_t r = i / KF;
    W0bf[i] = (k < FF) ? (__bf16)W0[r * FF + k] : (__bf16)0.0f;
  }
  const size_t NU = (size_t)G2 * HH;
  for (size_t i = tid; i < NU; i += stride) {
    U0bf[i] = (__bf16)U0[i];
    W1bf[i] = (__bf16)W1[i];
    U1bf[i] = (__bf16)U1[i];
  }
  for (size_t i = tid; i < (size_t)2 * BB * HH; i += stride) {
    hx0[i] = (__bf16)0.0f;
    hx1[i] = (__bf16)0.0f;
  }
  if (tid < 2) bar[tid] = 0u;
}

// ------------------------------------- pre-GEMM + LayerNorm (both layers)
// One WG per 16-row M-tile (24000 rows -> 1500 WGs). 8 waves, each wave does
// 8 N-tiles (128 gate columns). Cross-iteration software pipeline: fragments
// for k+1 are prefetched (pinned by sched_barrier) while k's WMMAs execute.
__global__ __launch_bounds__(256)
void k_gemm_ln(const __bf16* __restrict__ A, int K,
               const __bf16* __restrict__ Wm,
               const float* __restrict__ gamma, const float* __restrict__ beta,
               __bf16* __restrict__ wout) {
  __shared__ float wsm[16 * G2];           // 64 KB
  const int tid  = threadIdx.x;
  const int lane = tid & 31;
  const int wv   = tid >> 5;               // 0..7
  const int l    = lane & 15;
  const int hi   = lane >> 4;
  const int mtile = blockIdx.x;

  const __bf16* arow = A + (size_t)(mtile * 16 + l) * K;
  const __bf16* browp[8];
#pragma unroll
  for (int n = 0; n < 8; ++n)
    browp[n] = Wm + (size_t)((wv * 8 + n) * 16 + l) * K;

  v8f acc[8];
#pragma unroll
  for (int n = 0; n < 8; ++n) acc[n] = vzero8();

  const int ksteps = K >> 5;
  v16bf aCur = load_a_frag(arow, 0);
  v16bf bCur[8];
#pragma unroll
  for (int n = 0; n < 8; ++n) bCur[n] = load_b_frag(browp[n], 0);

#pragma unroll 2
  for (int k = 0; k < ksteps; ++k) {
    const int kn = (k + 1 < ksteps) ? (k + 1) * 32 : k * 32;
    v16bf aN = load_a_frag(arow, kn);
    v16bf bN[8];
#pragma unroll
    for (int n = 0; n < 8; ++n) bN[n] = load_b_frag(browp[n], kn);
    __builtin_amdgcn_sched_barrier(0);   // keep next-step prefetch above WMMAs
#pragma unroll
    for (int n = 0; n < 8; ++n) acc[n] = wmma_bf16(aCur, bCur[n], acc[n]);
    aCur = aN;
#pragma unroll
    for (int n = 0; n < 8; ++n) bCur[n] = bN[n];
  }
  // C layout: VGPR i -> row i + 8*hi, col = lane&15
#pragma unroll
  for (int n = 0; n < 8; ++n)
#pragma unroll
    for (int i = 0; i < 8; ++i)
      wsm[(i + 8 * hi) * G2 + (wv * 8 + n) * 16 + l] = acc[n][i];
  __syncthreads();

  // LayerNorm: 16 threads per row (rows are half-waves -> shfl width 16)
  const int m = tid >> 4;
  const int j = tid & 15;
  float s = 0.0f, q = 0.0f;
  for (int c = j; c < G2; c += 16) {
    float v = wsm[m * G2 + c];
    s += v; q += v * v;
  }
#pragma unroll
  for (int o = 1; o < 16; o <<= 1) {
    s += __shfl_xor(s, o, 16);
    q += __shfl_xor(q, o, 16);
  }
  float mu  = s * (1.0f / G2);
  float var = q * (1.0f / G2) - mu * mu;
  var = var > 0.0f ? var : 0.0f;
  float rs = rsqrtf(var + 1e-5f);
  const size_t rowoff = (size_t)(mtile * 16 + m) * G2;
  for (int c = j * 64; c < j * 64 + 64; ++c) {
    float v = (wsm[m * G2 + c] - mu) * rs * gamma[c] + beta[c];
    wout[rowoff + c] = (__bf16)v;
  }
}

// ----------------------------------------------------- grid barrier
__device__ __forceinline__ void grid_barrier(unsigned* cnt, unsigned* gen,
                                             unsigned nwg) {
  __threadfence();                 // release h stores to agent scope
  __syncthreads();
  if (threadIdx.x == 0) {
    unsigned g0 = __hip_atomic_load(gen, __ATOMIC_RELAXED, __HIP_MEMORY_SCOPE_AGENT);
    unsigned a  = __hip_atomic_fetch_add(cnt, 1u, __ATOMIC_ACQ_REL, __HIP_MEMORY_SCOPE_AGENT);
    if (a == nwg - 1u) {
      __hip_atomic_store(cnt, 0u, __ATOMIC_RELAXED, __HIP_MEMORY_SCOPE_AGENT);
      __hip_atomic_fetch_add(gen, 1u, __ATOMIC_RELEASE, __HIP_MEMORY_SCOPE_AGENT);
    } else {
      while (__hip_atomic_load(gen, __ATOMIC_ACQUIRE, __HIP_MEMORY_SCOPE_AGENT) == g0)
        __builtin_amdgcn_s_sleep(2);
    }
  }
  __syncthreads();
  __builtin_amdgcn_fence(__ATOMIC_ACQUIRE, "agent");   // invalidate WGP$ before next h read
}

// --------------------------------------------- persistent recurrence
// 8 WGs x 8 waves. WG g owns h columns [64g,64g+64): waves 0-3 compute the
// a-gate tiles, waves 4-7 the z-gate tiles. U slice (16x512 bf16 per wave)
// lives in 128 VGPRs. Per step: prefetch w biases, A fragments pipelined in
// 4-chunks with sched_barrier pinning each prefetch above the WMMA block.
__global__ __launch_bounds__(256)
void k_rec(const __bf16* __restrict__ U, const __bf16* __restrict__ wbuf,
           __bf16* __restrict__ hx,            // double-buffered 2 x [16][512]
           __bf16* __restrict__ hout_bf,       // layer0: bf16 output (A for layer1)
           float*  __restrict__ hout_f32,      // layer1: final f32 output
           unsigned* __restrict__ bar) {
  __shared__ float zs[16 * 64];
  __builtin_amdgcn_fence(__ATOMIC_ACQUIRE, "agent");

  const int tid  = threadIdx.x;
  const int lane = tid & 31;
  const int wv   = tid >> 5;         // 0..7
  const int l    = lane & 15;
  const int hi   = lane >> 4;
  const int g    = blockIdx.x;       // 0..7
  const bool isZ = (wv >= 4);
  const int nt   = isZ ? (32 + 4 * g + (wv - 4)) : (4 * g + wv);
  const int colg = nt * 16 + l;      // gate column owned by this lane

  // U tile resident in registers: 16 k-steps x 8 VGPRs = 128 VGPRs
  const __bf16* brow = U + (size_t)colg * HH;
  v16bf breg[16];
#pragma unroll
  for (int k = 0; k < 16; ++k) breg[k] = load_b_frag(brow, k * 32);

  // per-i w row pointers (advance by G2 per t)
  const __bf16* wrow[8];
#pragma unroll
  for (int i = 0; i < 8; ++i) {
    int m = i + 8 * hi;
    wrow[i] = wbuf + (size_t)m * TT * G2 + colg;
  }
  const __bf16* arow0 = hx + (size_t)l * HH;              // buffer 0
  const __bf16* arow1 = arow0 + (size_t)BB * HH;          // buffer 1

  for (int t = 0; t < TT; ++t) {
    const __bf16* arow = (t & 1) ? arow1 : arow0;
    const __bf16* hr   = (t & 1) ? (hx + BB * HH) : hx;
    __bf16*       hw   = (t & 1) ? hx : (hx + BB * HH);

    // prefetch gate biases for this step (in flight during WMMA chain)
    float wreg[8];
#pragma unroll
    for (int i = 0; i < 8; ++i)
      wreg[i] = (float)wrow[i][(size_t)t * G2];

    // software-pipelined h @ U^T: chunks of 4 k-steps
    v8f acc = vzero8();
    v16bf a0 = load_a_frag(arow, 0);
    v16bf a1 = load_a_frag(arow, 32);
    v16bf a2 = load_a_frag(arow, 64);
    v16bf a3 = load_a_frag(arow, 96);
#pragma unroll
    for (int c = 0; c < 4; ++c) {
      v16bf c0 = a0, c1 = a1, c2 = a2, c3 = a3;
      if (c < 3) {
        const int kb = (c + 1) * 128;
        a0 = load_a_frag(arow, kb);
        a1 = load_a_frag(arow, kb + 32);
        a2 = load_a_frag(arow, kb + 64);
        a3 = load_a_frag(arow, kb + 96);
      }
      __builtin_amdgcn_sched_barrier(0); // pin chunk c+1 loads above chunk c WMMAs
      acc = wmma_bf16(c0, breg[c * 4 + 0], acc);
      acc = wmma_bf16(c1, breg[c * 4 + 1], acc);
      acc = wmma_bf16(c2, breg[c * 4 + 2], acc);
      acc = wmma_bf16(c3, breg[c * 4 + 3], acc);
    }

    if (isZ) {  // sigmoid gate -> LDS
#pragma unroll
      for (int i = 0; i < 8; ++i) {
        int m = i + 8 * hi;                       // batch row
        float zv = 1.0f / (1.0f + __expf(-(acc[i] + wreg[i])));
        zs[m * 64 + 16 * (wv - 4) + l] = zv;
      }
    }
    __syncthreads();
    if (!isZ) { // relu gate + blend + state/output stores
#pragma unroll
      for (int i = 0; i < 8; ++i) {
        int m = i + 8 * hi;
        float av = acc[i] + wreg[i]; av = av > 0.0f ? av : 0.0f;
        float zv = zs[m * 64 + 16 * wv + l];
        int   hc = 64 * g + 16 * wv + l;
        float ho = (float)hr[(size_t)m * HH + hc];
        float hn = zv * ho + (1.0f - zv) * av;
        hw[(size_t)m * HH + hc] = (__bf16)hn;
        size_t orow = (size_t)(m * TT + t) * HH + hc;
        if (hout_bf) hout_bf[orow] = (__bf16)hn;
        else         hout_f32[orow] = hn;
      }
    }
    grid_barrier(bar, bar + 1, gridDim.x);
  }
}

// ---------------------------------------------------------------- launch
extern "C" void kernel_launch(void* const* d_in, const int* in_sizes, int n_in,
                              void* d_out, int out_size, void* d_ws, size_t ws_size,
                              hipStream_t stream) {
  (void)in_sizes; (void)n_in; (void)out_size; (void)ws_size;
  const float* x  = (const float*)d_in[0];
  const float* W0 = (const float*)d_in[1];
  const float* U0 = (const float*)d_in[2];
  const float* g0 = (const float*)d_in[3];
  const float* b0 = (const float*)d_in[4];
  const float* W1 = (const float*)d_in[5];
  const float* U1 = (const float*)d_in[6];
  const float* g1 = (const float*)d_in[7];
  const float* b1 = (const float*)d_in[8];

  char* ws = (char*)d_ws;
  __bf16* wbuf  = (__bf16*)(ws + OFF_WBUF);
  __bf16* h0buf = (__bf16*)(ws + OFF_H0);
  __bf16* xbf   = (__bf16*)(ws + OFF_XBF);
  __bf16* W0bf  = (__bf16*)(ws + OFF_W0BF);
  __bf16* U0bf  = (__bf16*)(ws + OFF_U0BF);
  __bf16* W1bf  = (__bf16*)(ws + OFF_W1BF);
  __bf16* U1bf  = (__bf16*)(ws + OFF_U1BF);
  __bf16* hx0   = (__bf16*)(ws + OFF_HX0);
  __bf16* hx1   = (__bf16*)(ws + OFF_HX1);
  unsigned* bar = (unsigned*)(ws + OFF_BAR);

  k_prep<<<768, 256, 0, stream>>>(x, W0, U0, W1, U1, xbf, W0bf, U0bf, W1bf,
                                  U1bf, hx0, hx1, bar);
  // layer 0
  k_gemm_ln<<<1500, 256, 0, stream>>>(xbf, KF, W0bf, g0, b0, wbuf);
  k_rec<<<8, 256, 0, stream>>>(U0bf, wbuf, hx0, h0buf, nullptr, bar);
  // layer 1
  k_gemm_ln<<<1500, 256, 0, stream>>>(h0buf, HH, W1bf, g1, b1, wbuf);
  k_rec<<<8, 256, 0, stream>>>(U1bf, wbuf, hx1, nullptr, (float*)d_out, bar);
}
